// FCGHPN_59794534695173
// MI455X (gfx1250) — compile-verified, matrix operand order checked by reading, not run on hardware
//
#include <hip/hip_runtime.h>
#include <hip/hip_bf16.h>

typedef __attribute__((ext_vector_type(16))) _Float16 v16h;
typedef __attribute__((ext_vector_type(8)))  float    v8f;

#define B_    4096
#define N_    66
#define FIN_  64
#define HID_  128
#define GOUT_ 56
#define T2V_  8
#define K_    10
#define H_    4
#define DH_   16
#define NP_   80     // node count padded to 5 M-tiles
#define KP_   96     // node count padded to 3 K-steps of 32
#define NWAVES 8
#define TPB   256

__device__ __forceinline__ float wred_max(float v) {
  #pragma unroll
  for (int o = 16; o > 0; o >>= 1) v = fmaxf(v, __shfl_xor(v, o, 32));
  return v;
}
__device__ __forceinline__ float wred_sum(float v) {
  #pragma unroll
  for (int o = 16; o > 0; o >>= 1) v += __shfl_xor(v, o, 32);
  return v;
}

// A fragment (16x32 f16): lanes 0-15 rows; halves 0-7 -> K 0-7 (lo lanes) / 8-15 (hi lanes),
// halves 8-15 -> K 16-23 / 24-31 (per CDNA5 ISA 7.12.2).
__device__ __forceinline__ v16h load_frag_a(const _Float16* p, int lda, int row0, int k0) {
  const int lane = threadIdx.x & 31;
  const int hi = lane >> 4;
  const _Float16* base = p + (row0 + (lane & 15)) * lda + k0;
  v16h a;
  #pragma unroll
  for (int i = 0; i < 8; ++i) a[i] = base[hi * 8 + i];
  #pragma unroll
  for (int i = 0; i < 8; ++i) a[8 + i] = base[16 + hi * 8 + i];
  return a;
}

// B fragment (32x16 f16), B stored row-major K x ldb: lane = column, same K striping as A.
__device__ __forceinline__ v16h load_frag_b(const _Float16* p, int ldb, int k0, int col0) {
  const int lane = threadIdx.x & 31;
  const int hi = lane >> 4;
  const _Float16* base = p + k0 * ldb + col0 + (lane & 15);
  v16h b;
  #pragma unroll
  for (int i = 0; i < 8; ++i) b[i] = base[(hi * 8 + i) * ldb];
  #pragma unroll
  for (int i = 0; i < 8; ++i) b[8 + i] = base[(16 + hi * 8 + i) * ldb];
  return b;
}

// All 8 waves cooperate on an Mt x Nt tile grid, K-loop of 32 per WMMA.
template <class InitF, class EpiF>
__device__ __forceinline__ void gemm_f16(const _Float16* A, int lda,
                                         const _Float16* Bm, int ldb,
                                         int Mt, int Nt, int K,
                                         InitF init, EpiF epi) {
  const int lane = threadIdx.x & 31;
  const int wave = threadIdx.x >> 5;
  const int hi = lane >> 4, lo = lane & 15;
  for (int t = wave; t < Mt * Nt; t += NWAVES) {
    const int tm = t / Nt, tn = t % Nt;
    const int col = tn * 16 + lo;
    v8f acc;
    #pragma unroll
    for (int r = 0; r < 8; ++r) acc[r] = init(col);
    for (int k0 = 0; k0 < K; k0 += 32) {
      v16h a = load_frag_a(A, lda, tm * 16, k0);
      v16h b = load_frag_b(Bm, ldb, k0, tn * 16);
      acc = __builtin_amdgcn_wmma_f32_16x16x32_f16(false, a, false, b,
                                                   (short)0, acc, false, false);
    }
    #pragma unroll
    for (int r = 0; r < 8; ++r) epi(tm * 16 + r + 8 * hi, col, acc[r]);
  }
}

__global__ __launch_bounds__(TPB)
void fused_graph_kernel(const float* __restrict__ adj, const float* __restrict__ inten,
                        const float* __restrict__ xin, const float* __restrict__ t2v,
                        const int*  __restrict__ nbrs,
                        const float* __restrict__ W1, const float* __restrict__ b1,
                        const float* __restrict__ W2, const float* __restrict__ b2,
                        const float* __restrict__ Wq, const float* __restrict__ bq,
                        const float* __restrict__ Wk, const float* __restrict__ bk,
                        const float* __restrict__ Wv, const float* __restrict__ bv,
                        const float* __restrict__ Wo, const float* __restrict__ bo,
                        const float* __restrict__ Win, const float* __restrict__ Wout,
                        float* __restrict__ o_emb, float* __restrict__ o_od,
                        float* __restrict__ o_ind, float* __restrict__ o_outd) {
  __shared__ _Float16 sAt[NP_ * KP_];    // binary A transposed, f16 {0,1}
  __shared__ _Float16 sX [KP_ * 64];     // hn, later q_in
  __shared__ _Float16 sZ [KP_ * 128];    // relu(h1)*n_out (K-padded)
  __shared__ _Float16 sY [NP_ * 128];    // GCN aggregation results
  __shared__ _Float16 sW [8192];         // current weight matrix, f16
  __shared__ float    sBias[128];
  __shared__ _Float16 sq[NP_ * 64], sk[NP_ * 64], sv[NP_ * 64];
  __shared__ float    sWo[64];
  __shared__ int      sNbr[N_ * K_];
  __shared__ float    sDegOut[NP_], sNout[NP_], sNin[NP_];
  __shared__ float    sEmb[NP_], sE[NP_], sSig[NP_];

  const int b    = blockIdx.x;
  const int tid  = threadIdx.x;
  const int lane = tid & 31;
  const int wave = tid >> 5;

  // warm shared weights into cache (global_prefetch_b8)
  __builtin_prefetch(W1, 0, 2);
  __builtin_prefetch(W2, 0, 2);
  __builtin_prefetch(Win, 0, 2);
  __builtin_prefetch(Wout, 0, 2);

  for (int i = tid; i < NP_ * KP_;  i += TPB) sAt[i] = (_Float16)0.f;
  for (int i = tid; i < KP_ * 64;   i += TPB) sX[i]  = (_Float16)0.f;
  for (int i = tid; i < KP_ * 128;  i += TPB) sZ[i]  = (_Float16)0.f;
  for (int i = tid; i < N_ * K_;    i += TPB) sNbr[i] = nbrs[i];
  if (tid < 64) sWo[tid] = Wo[tid];
  __syncthreads();

  // ---- Phase 0: row softmax + threshold -> binary A, stored transposed; deg_out ----
  const float* adjb = adj + (size_t)b * N_ * N_;
  for (int i = wave; i < N_; i += NWAVES) {
    const int j0 = lane, j1 = lane + 32, j2 = lane + 64;
    float x0 = (j0 < N_) ? adjb[i * N_ + j0] : -1e30f;
    float x1 = (j1 < N_) ? adjb[i * N_ + j1] : -1e30f;
    float x2 = (j2 < N_) ? adjb[i * N_ + j2] : -1e30f;
    float m  = wred_max(fmaxf(x0, fmaxf(x1, x2)));
    float e0 = (j0 < N_) ? __expf(x0 - m) : 0.f;
    float e1 = (j1 < N_) ? __expf(x1 - m) : 0.f;
    float e2 = (j2 < N_) ? __expf(x2 - m) : 0.f;
    float s  = wred_sum(e0 + e1 + e2);
    float thr = 1e-8f * s;     // e/s >= 1e-8  <=>  e >= thr
    float deg = 0.f;
    if (j0 < N_) { float a = (e0 >= thr) ? 1.f : 0.f; sAt[j0 * KP_ + i] = (_Float16)a; deg += a; }
    if (j1 < N_) { float a = (e1 >= thr) ? 1.f : 0.f; sAt[j1 * KP_ + i] = (_Float16)a; deg += a; }
    if (j2 < N_) { float a = (e2 >= thr) ? 1.f : 0.f; sAt[j2 * KP_ + i] = (_Float16)a; deg += a; }
    deg = wred_sum(deg);
    if (lane == 0) sDegOut[i] = deg;
  }
  if (tid >= N_ && tid < NP_) sDegOut[tid] = 0.f;
  __syncthreads();

  // ---- Phase 0b: deg_in = column sums of A = row sums of At; normalizers ----
  for (int j = wave; j < N_; j += NWAVES) {
    float s = (float)sAt[j * KP_ + lane] + (float)sAt[j * KP_ + lane + 32]
            + (float)sAt[j * KP_ + lane + 64];
    s = wred_sum(s);
    if (lane == 0) sNin[j] = (s > 0.f) ? rsqrtf(s) : 0.f;
  }
  if (tid < NP_) sNout[tid] = (sDegOut[tid] > 0.f) ? rsqrtf(sDegOut[tid]) : 0.f;
  if (tid >= N_ && tid < NP_) sNin[tid] = 0.f;
  __syncthreads();

  // ---- Phase 1: hn = x * n_out ; stage W1/b1 ----
  const float* xb = xin + (size_t)b * N_ * FIN_;
  for (int idx = tid; idx < N_ * FIN_; idx += TPB) {
    int i = idx >> 6, f = idx & 63;
    sX[i * 64 + f] = (_Float16)(xb[idx] * sNout[i]);
  }
  for (int idx = tid; idx < 64 * 128; idx += TPB) sW[idx] = (_Float16)W1[idx];
  if (tid < 128) sBias[tid] = b1[tid];
  __syncthreads();

  // ---- GCN1 aggregate: m = At(80x96) @ hn(96x64), scale n_in ----
  gemm_f16(sAt, KP_, sX, 64, NP_ / 16, 64 / 16, KP_,
           [&](int) { return 0.f; },
           [&](int r, int c, float v) { sY[r * 64 + c] = (_Float16)(v * sNin[r]); });
  __syncthreads();

  // ---- GCN1 dense: h1 = relu(Y @ W1 + b1); fold *n_out for layer 2 ----
  gemm_f16(sY, 64, sW, 128, NP_ / 16, 128 / 16, 64,
           [&](int c) { return sBias[c]; },
           [&](int r, int c, float v) { sZ[r * 128 + c] = (_Float16)(fmaxf(v, 0.f) * sNout[r]); });
  __syncthreads();

  // ---- GCN2 aggregate (overlap W2 staging; sW/sBias idle here) ----
  for (int idx = tid; idx < 128 * 64; idx += TPB) {
    int r = idx >> 6, c = idx & 63;
    sW[idx] = (_Float16)((c < GOUT_) ? W2[r * GOUT_ + c] : 0.f);
  }
  if (tid < 64) sBias[tid] = (tid < GOUT_) ? b2[tid] : 0.f;
  gemm_f16(sAt, KP_, sZ, 128, NP_ / 16, 128 / 16, KP_,
           [&](int) { return 0.f; },
           [&](int r, int c, float v) { sY[r * 128 + c] = (_Float16)(v * sNin[r]); });
  __syncthreads();

  // ---- GCN2 dense: h2 = Y(80x128) @ W2(128x64 padded) + b2 -> q_in cols 0..55 ----
  gemm_f16(sY, 128, sW, 64, NP_ / 16, 64 / 16, 128,
           [&](int c) { return sBias[c]; },
           [&](int r, int c, float v) { sX[r * 64 + c] = (_Float16)v; });
  __syncthreads();

  // splice time2vec into q_in cols 56..63
  const float* tb = t2v + (size_t)b * N_ * T2V_;
  for (int idx = tid; idx < N_ * T2V_; idx += TPB) {
    int i = idx / T2V_, t = idx % T2V_;
    sX[i * 64 + GOUT_ + t] = (_Float16)tb[idx];
  }
  __syncthreads();

  // ---- Q projection: q_in(80x64) @ Wq(64x64) + bq ----
  for (int idx = tid; idx < 64 * 64; idx += TPB) sW[idx] = (_Float16)Wq[idx];
  if (tid < 64) sBias[tid] = bq[tid];
  __syncthreads();
  gemm_f16(sX, 64, sW, 64, NP_ / 16, 64 / 16, 64,
           [&](int c) { return sBias[c]; },
           [&](int r, int c, float v) { sq[r * 64 + c] = (_Float16)v; });
  __syncthreads();

  // ---- K projection ----
  for (int idx = tid; idx < 64 * 64; idx += TPB) sW[idx] = (_Float16)Wk[idx];
  if (tid < 64) sBias[tid] = bk[tid];
  __syncthreads();
  gemm_f16(sX, 64, sW, 64, NP_ / 16, 64 / 16, 64,
           [&](int c) { return sBias[c]; },
           [&](int r, int c, float v) { sk[r * 64 + c] = (_Float16)v; });
  __syncthreads();

  // ---- V projection ----
  for (int idx = tid; idx < 64 * 64; idx += TPB) sW[idx] = (_Float16)Wv[idx];
  if (tid < 64) sBias[tid] = bv[tid];
  __syncthreads();
  gemm_f16(sX, 64, sW, 64, NP_ / 16, 64 / 16, 64,
           [&](int c) { return sBias[c]; },
           [&](int r, int c, float v) { sv[r * 64 + c] = (_Float16)v; });
  __syncthreads();

  // ---- Attention (per node/head) fused with @Wo reduction ----
  if (tid < NP_) sEmb[tid] = bo[0];
  __syncthreads();
  const float* intb = inten + (size_t)b * N_ * K_;
  for (int t = tid; t < N_ * H_; t += TPB) {
    const int n = t / H_, h = t % H_;
    float qv[DH_];
    #pragma unroll
    for (int d = 0; d < DH_; ++d) qv[d] = (float)sq[n * 64 + h * DH_ + d];
    float sc[K_];
    float mx = -1e30f;
    #pragma unroll
    for (int k = 0; k < K_; ++k) {
      const int nb = sNbr[n * K_ + k];
      float dot = 0.f;
      #pragma unroll
      for (int d = 0; d < DH_; ++d) dot += qv[d] * (float)sk[nb * 64 + h * DH_ + d];
      float s = dot * 0.25f * intb[n * K_ + k];  // 1/sqrt(16) = 0.25
      sc[k] = s;
      mx = fmaxf(mx, s);
    }
    float sum = 0.f;
    #pragma unroll
    for (int k = 0; k < K_; ++k) { sc[k] = __expf(sc[k] - mx); sum += sc[k]; }
    const float inv = 1.f / sum;
    float part = 0.f;
    #pragma unroll
    for (int k = 0; k < K_; ++k) {
      const int nb = sNbr[n * K_ + k];
      const float w = sc[k] * inv;
      #pragma unroll
      for (int d = 0; d < DH_; ++d)
        part += w * (float)sv[nb * 64 + h * DH_ + d] * sWo[h * DH_ + d];
    }
    atomicAdd(&sEmb[n], part);   // ds_add_f32
  }
  __syncthreads();

  // ---- embedding out + sigmoid staging ----
  float* ob = o_emb + (size_t)b * N_;
  for (int n = tid; n < N_; n += TPB) {
    float e = sEmb[n];
    ob[n] = e;
    sE[n] = e;
    sSig[n] = 1.f / (1.f + __expf(-e));
  }
  __syncthreads();

  // ---- OD matrix (rank-1 outer product) ----
  float* odb = o_od + (size_t)b * N_ * N_;
  for (int idx = tid; idx < N_ * N_; idx += TPB) {
    int i = idx / N_, j = idx % N_;
    odb[idx] = sSig[i] * sE[j];
  }

  // ---- degree products: e @ weight_in / weight_out ----
  for (int t = tid; t < 2 * N_; t += TPB) {
    const int col = (t < N_) ? t : (t - N_);
    const float* Wm = (t < N_) ? Win : Wout;
    float acc = 0.f;
    for (int i = 0; i < N_; ++i) acc += sE[i] * Wm[i * N_ + col];
    if (t < N_) o_ind[(size_t)b * N_ + col] = acc;
    else        o_outd[(size_t)b * N_ + col] = acc;
  }
}

extern "C" void kernel_launch(void* const* d_in, const int* in_sizes, int n_in,
                              void* d_out, int out_size, void* d_ws, size_t ws_size,
                              hipStream_t stream) {
  (void)in_sizes; (void)n_in; (void)out_size; (void)d_ws; (void)ws_size;
  const float* adj   = (const float*)d_in[0];
  const float* inten = (const float*)d_in[1];
  const float* xin   = (const float*)d_in[2];
  const float* t2v   = (const float*)d_in[3];
  const int*   nbr   = (const int*)d_in[4];
  const float* W1 = (const float*)d_in[5];
  const float* b1 = (const float*)d_in[6];
  const float* W2 = (const float*)d_in[7];
  const float* b2 = (const float*)d_in[8];
  const float* Wq = (const float*)d_in[9];
  const float* bq = (const float*)d_in[10];
  const float* Wk = (const float*)d_in[11];
  const float* bk = (const float*)d_in[12];
  const float* Wv = (const float*)d_in[13];
  const float* bv = (const float*)d_in[14];
  const float* Wo = (const float*)d_in[15];
  const float* bo = (const float*)d_in[16];
  const float* Win  = (const float*)d_in[17];
  const float* Wout = (const float*)d_in[18];

  float* out    = (float*)d_out;
  float* o_emb  = out;                                   // (B,N,1)
  float* o_od   = o_emb + (size_t)B_ * N_;               // (B,N,N)
  float* o_ind  = o_od  + (size_t)B_ * N_ * N_;          // (B,N)
  float* o_outd = o_ind + (size_t)B_ * N_;               // (B,N)

  hipLaunchKernelGGL(fused_graph_kernel, dim3(B_), dim3(TPB), 0, stream,
                     adj, inten, xin, t2v, nbr,
                     W1, b1, W2, b2, Wq, bq, Wk, bk, Wv, bv, Wo, bo,
                     Win, Wout, o_emb, o_od, o_ind, o_outd);
}
